// QuantumGeometricAttention_824633721118
// MI455X (gfx1250) — matrix operand order repaired and checked by reading, not compiled, not run on hardware
//
#include <hip/hip_runtime.h>
#include <stdint.h>
#include <stddef.h>

// ---------------------------------------------------------------------------
// Problem constants (B=4, S=1024, D=256, H=8, L=3)
// ---------------------------------------------------------------------------
#define CB 4
#define CS 1024
#define CD 256
#define CH 8
#define CHD 2048          // H*D
#define CM 4096           // B*S
#define CNQKV 6144        // 3*H*D

typedef __bf16 bf16_t;
typedef bf16_t bf16x16 __attribute__((ext_vector_type(16)));
typedef float  v8f     __attribute__((ext_vector_type(8)));
typedef unsigned int u32x4 __attribute__((ext_vector_type(4)));

#define DEVINL __device__ __forceinline__

// ---- CDNA5 async global->LDS path (guarded; falls back to VGPR staging) ----
#if __has_builtin(__builtin_amdgcn_global_load_async_to_lds_b128)
#define USE_ASYNC_LDS 1
typedef int vs4i __attribute__((vector_size(16)));
typedef __attribute__((address_space(1))) vs4i as1_vs4i;
typedef __attribute__((address_space(3))) vs4i as3_vs4i;
#endif

DEVINL void ld16(void* lds_dst, const void* gsrc) {
#ifdef USE_ASYNC_LDS
  __builtin_amdgcn_global_load_async_to_lds_b128(
      (as1_vs4i*)(unsigned long long)gsrc,
      (as3_vs4i*)(unsigned int)(unsigned long long)lds_dst, 0, 0);
#else
  *(u32x4*)lds_dst = *(const u32x4*)gsrc;
#endif
}

DEVINL void wait_async_lds() {
#ifdef USE_ASYNC_LDS
#if __has_builtin(__builtin_amdgcn_s_wait_asynccnt)
  __builtin_amdgcn_s_wait_asynccnt(0);
#else
  asm volatile("s_wait_asynccnt 0x0" ::: "memory");
#endif
#endif
}

DEVINL unsigned short f2bf(float f) {
  union { float f; unsigned int u; } x; x.f = f;
  unsigned int u = x.u;
  return (unsigned short)((u + 0x7FFFu + ((u >> 16) & 1u)) >> 16);  // RNE
}

DEVINL v8f zero8() {
  v8f z;
#pragma unroll
  for (int i = 0; i < 8; ++i) z[i] = 0.f;
  return z;
}

union FragU { bf16x16 v; u32x4 q[2]; };

// A fragment (16x32 bf16, M x K): lane half 0 holds K {0..7,16..23}, half 1 holds {8..15,24..31}
DEVINL bf16x16 frag_a(const unsigned short* base, int row, int ld, int kh) {
  FragU u;
  u.q[0] = *(const u32x4*)(base + row * ld + kh * 8);
  u.q[1] = *(const u32x4*)(base + row * ld + 16 + kh * 8);
  return u.v;
}
// B fragment (32x16 bf16, K x N) from row-major [N,K] storage: lane half h holds K {8h..8h+7,16+8h..16+8h+7}
DEVINL bf16x16 frag_b(const unsigned short* base, int row, int ld, int kh) {
  FragU u;
  u.q[0] = *(const u32x4*)(base + row * ld + kh * 8);
  u.q[1] = *(const u32x4*)(base + row * ld + 16 + kh * 8);
  return u.v;
}

DEVINL v8f wmma_bf16(bf16x16 a, bf16x16 b, v8f c) {
  return __builtin_amdgcn_wmma_f32_16x16x32_bf16(
      /*neg_a=*/false, a, /*neg_b=*/false, b,
      /*c_mod=*/(short)0, c, /*reuse_a=*/false, /*reuse_b=*/false);
}

// ---------------------------------------------------------------------------
// Generic complex WMMA GEMM:  C = A @ B^T  (A:[M,K], B:[N,K] row-major, K-major)
// Block tile 64(M) x 128(N) x 32(K), 8 waves, each wave owns 32x32 of C.
// MODE 0: full complex   Cr = rr-ii, Ci = ri+ir        (4 WMMA / tile / step)
// MODE 1: Re(A·conj(B))  Cr = (rr+ii)*scale            (2 WMMA)
// MODE 2: real A x cplx B Cr = A·Br, Ci = A·Bi         (2 WMMA)
// EPI 0: QKV scatter (q/k [B,H,S,D] bf16, v transposed [B,H,D,S] bf16, +bias)
// EPI 1: scores f32 [z][S][S];  EPI 2: PV -> attn bf16 [B,S,H*D]
// EPI 3: bias bf16 r/i;  EPI 4: bias+CReLU bf16 r/i;  EPI 5: bias+CReLU f32 x2
// ---------------------------------------------------------------------------
#define BM 64
#define BN 128
#define BKT 32
#define LSTR 40   // LDS row stride in bf16 elems (32 + 8 pad)

struct EpiArgs {
  void* o0; void* o1; void* o2; void* o3; void* o4; void* o5;
  const float* biasR; const float* biasI;
  float scale;
};

template <int MODE, int EPI>
__global__ __launch_bounds__(256) void cgemm_wmma_kernel(
    const unsigned short* __restrict__ Ar, const unsigned short* __restrict__ Ai,
    const unsigned short* __restrict__ Br, const unsigned short* __restrict__ Bi,
    int K, int lda, int ldb, long long sAz, long long sBz, EpiArgs ea) {
  __shared__ __align__(16) unsigned short shAr[2][BM * LSTR];
  __shared__ __align__(16) unsigned short shAi[2][BM * LSTR];
  __shared__ __align__(16) unsigned short shBr[2][BN * LSTR];
  __shared__ __align__(16) unsigned short shBi[2][BN * LSTR];

  const int t  = threadIdx.x;
  const int z  = blockIdx.z;
  const int m0 = blockIdx.y * BM;
  const int n0 = blockIdx.x * BN;

  Ar += (long long)z * sAz;
  if (MODE != 2) Ai += (long long)z * sAz;
  Br += (long long)z * sBz;
  Bi += (long long)z * sBz;

  // cooperative tile-load mapping (16B per thread-slot)
  const int lrow = t >> 2;           // A: 0..63
  const int lcol = (t & 3) << 3;

  // wave mapping: 8 waves -> 2 (M-groups of 32) x 4 (N-groups of 32)
  const int lane   = t & 31;
  const int wv     = t >> 5;
  const int wm     = wv & 1;
  const int wn     = wv >> 1;
  const int lrow16 = lane & 15;
  const int kh     = lane >> 4;

  auto load_tiles = [&](int buf, int k0) {
    ld16(&shAr[buf][lrow * LSTR + lcol],
         Ar + (size_t)(m0 + lrow) * lda + k0 + lcol);
    if (MODE != 2)
      ld16(&shAi[buf][lrow * LSTR + lcol],
           Ai + (size_t)(m0 + lrow) * lda + k0 + lcol);
#pragma unroll
    for (int ii = 0; ii < 2; ++ii) {
      const int idx  = t + (ii << 8);   // 0..511
      const int brow = idx >> 2;        // 0..127
      const int bcol = (idx & 3) << 3;
      ld16(&shBr[buf][brow * LSTR + bcol],
           Br + (size_t)(n0 + brow) * ldb + k0 + bcol);
      ld16(&shBi[buf][brow * LSTR + bcol],
           Bi + (size_t)(n0 + brow) * ldb + k0 + bcol);
    }
  };

  v8f acc0[4], acc1[4], acc2[4], acc3[4];   // index = im*2 + j
#pragma unroll
  for (int q = 0; q < 4; ++q) {
    acc0[q] = zero8(); acc1[q] = zero8(); acc2[q] = zero8(); acc3[q] = zero8();
  }

  const int nsteps = K / BKT;
  load_tiles(0, 0);
  wait_async_lds();
  __syncthreads();

  for (int s = 0; s < nsteps; ++s) {
    const int cur = s & 1;
    if (s + 1 < nsteps) load_tiles(cur ^ 1, (s + 1) * BKT);

    bf16x16 arf[2], aif[2];
#pragma unroll
    for (int im = 0; im < 2; ++im) {
      const int arow = wm * 32 + im * 16 + lrow16;
      arf[im] = frag_a(shAr[cur], arow, LSTR, kh);
      aif[im] = arf[im];
      if (MODE != 2) aif[im] = frag_a(shAi[cur], arow, LSTR, kh);
    }
#pragma unroll
    for (int j = 0; j < 2; ++j) {
      const int bn = wn * 32 + j * 16 + lrow16;
      bf16x16 br = frag_b(shBr[cur], bn, LSTR, kh);
      bf16x16 bi = frag_b(shBi[cur], bn, LSTR, kh);
#pragma unroll
      for (int im = 0; im < 2; ++im) {
        const int q = im * 2 + j;
        if (MODE == 0) {
          acc0[q] = wmma_bf16(arf[im], br, acc0[q]);  // rr
          acc1[q] = wmma_bf16(aif[im], bi, acc1[q]);  // ii
          acc2[q] = wmma_bf16(arf[im], bi, acc2[q]);  // ri
          acc3[q] = wmma_bf16(aif[im], br, acc3[q]);  // ir
        } else if (MODE == 1) {
          acc0[q] = wmma_bf16(arf[im], br, acc0[q]);  // qr·kr
          acc1[q] = wmma_bf16(aif[im], bi, acc1[q]);  // qi·ki
        } else {
          acc0[q] = wmma_bf16(arf[im], br, acc0[q]);  // P·Vr
          acc2[q] = wmma_bf16(arf[im], bi, acc2[q]);  // P·Vi
        }
      }
    }
    wait_async_lds();
    __syncthreads();
  }

  // ------------------------- epilogue -------------------------
#pragma unroll
  for (int im = 0; im < 2; ++im) {
    const int mlb = m0 + wm * 32 + im * 16 + kh * 8;  // first of this lane's 8 rows
#pragma unroll
    for (int j = 0; j < 2; ++j) {
      const int q  = im * 2 + j;
      const int ng = n0 + wn * 32 + j * 16 + lrow16;  // lane's column
      float cr[8], ci[8];
#pragma unroll
      for (int r = 0; r < 8; ++r) {
        if (MODE == 0)      { cr[r] = acc0[q][r] - acc1[q][r]; ci[r] = acc2[q][r] + acc3[q][r]; }
        else if (MODE == 1) { cr[r] = (acc0[q][r] + acc1[q][r]) * ea.scale; ci[r] = 0.f; }
        else                { cr[r] = acc0[q][r]; ci[r] = acc2[q][r]; }
      }

      if constexpr (EPI == 0) {  // QKV
        const float br_ = ea.biasR[ng], bi_ = ea.biasI[ng];
        const int part = ng >> 11;   // 0=q 1=k 2=v
        const int h    = (ng >> 8) & 7;
        const int d    = ng & 255;
        const int b    = mlb >> 10;
        const int s    = mlb & 1023;
        if (part == 2) {  // V transposed [B,H,D,S]: 8 consecutive s -> one b128 store
          unsigned short pr[8], pi[8];
#pragma unroll
          for (int r = 0; r < 8; ++r) { pr[r] = f2bf(cr[r] + br_); pi[r] = f2bf(ci[r] + bi_); }
          const size_t vb = (((size_t)(b * CH + h) * CD + d) * CS) + s;
          *(u32x4*)((unsigned short*)ea.o4 + vb) = *(const u32x4*)pr;
          *(u32x4*)((unsigned short*)ea.o5 + vb) = *(const u32x4*)pi;
        } else {
          unsigned short* dr = (unsigned short*)(part == 0 ? ea.o0 : ea.o2);
          unsigned short* di = (unsigned short*)(part == 0 ? ea.o1 : ea.o3);
          const size_t base = (((size_t)(b * CH + h) * CS + s) * CD) + d;
#pragma unroll
          for (int r = 0; r < 8; ++r) {
            dr[base + (size_t)r * CD] = f2bf(cr[r] + br_);
            di[base + (size_t)r * CD] = f2bf(ci[r] + bi_);
          }
        }
      } else if constexpr (EPI == 1) {  // scores f32
        float* o = (float*)ea.o0 + (size_t)z * CS * CS;
#pragma unroll
        for (int r = 0; r < 8; ++r) o[(size_t)(mlb + r) * CS + ng] = cr[r];
      } else if constexpr (EPI == 2) {  // PV -> attn_out [B,S,H*D]
        const int b = z >> 3, h = z & 7;
        unsigned short* orp = (unsigned short*)ea.o0;
        unsigned short* oip = (unsigned short*)ea.o1;
        const size_t base = ((size_t)(b * CS + mlb) * CHD) + h * CD + ng;
#pragma unroll
        for (int r = 0; r < 8; ++r) {
          orp[base + (size_t)r * CHD] = f2bf(cr[r]);
          oip[base + (size_t)r * CHD] = f2bf(ci[r]);
        }
      } else if constexpr (EPI == 3 || EPI == 4) {
        const float br_ = ea.biasR[ng], bi_ = ea.biasI[ng];
        unsigned short* orp = (unsigned short*)ea.o0;
        unsigned short* oip = (unsigned short*)ea.o1;
#pragma unroll
        for (int r = 0; r < 8; ++r) {
          float vr = cr[r] + br_, vi = ci[r] + bi_;
          if (EPI == 4) { vr = fmaxf(vr, 0.f); vi = fmaxf(vi, 0.f); }
          orp[(size_t)(mlb + r) * CD + ng] = f2bf(vr);
          oip[(size_t)(mlb + r) * CD + ng] = f2bf(vi);
        }
      } else {  // EPI == 5: final CReLU -> f32 interleaved [M,256,2]
        const float br_ = ea.biasR[ng], bi_ = ea.biasI[ng];
        float* o = (float*)ea.o0;
#pragma unroll
        for (int r = 0; r < 8; ++r) {
          float vr = fmaxf(cr[r] + br_, 0.f);
          float vi = fmaxf(ci[r] + bi_, 0.f);
          const size_t idx = ((size_t)(mlb + r) * CD + ng) * 2;
          o[idx] = vr;
          o[idx + 1] = vi;
        }
      }
    }
  }
}

// ---------------------------------------------------------------------------
// Row softmax over 1024 f32 -> bf16 P
// ---------------------------------------------------------------------------
__global__ __launch_bounds__(256) void softmax_kernel(const float* __restrict__ S_,
                                                      unsigned short* __restrict__ P) {
  __shared__ float red[256];
  const size_t row = blockIdx.x;
  const float* sp = S_ + row * CS;
  const int t = threadIdx.x;
  float v[4];
  float mx = -3.4e38f;
#pragma unroll
  for (int i = 0; i < 4; ++i) { v[i] = sp[t + i * 256]; mx = fmaxf(mx, v[i]); }
  red[t] = mx;
  __syncthreads();
  for (int s = 128; s > 0; s >>= 1) {
    if (t < s) red[t] = fmaxf(red[t], red[t + s]);
    __syncthreads();
  }
  mx = red[0];
  __syncthreads();
  float sum = 0.f;
#pragma unroll
  for (int i = 0; i < 4; ++i) { v[i] = __expf(v[i] - mx); sum += v[i]; }
  red[t] = sum;
  __syncthreads();
  for (int s = 128; s > 0; s >>= 1) {
    if (t < s) red[t] += red[t + s];
    __syncthreads();
  }
  const float inv = 1.f / red[0];
  unsigned short* pp = P + row * CS;
#pragma unroll
  for (int i = 0; i < 4; ++i) pp[t + i * 256] = f2bf(v[i] * inv);
}

// ---------------------------------------------------------------------------
// f32 -> bf16 cast
// ---------------------------------------------------------------------------
__global__ __launch_bounds__(256) void cast_bf16_kernel(const float* __restrict__ in,
                                                        unsigned short* __restrict__ out, int n) {
  int i = blockIdx.x * blockDim.x + threadIdx.x;
  const int stride = gridDim.x * blockDim.x;
  for (; i < n; i += stride) out[i] = f2bf(in[i]);
}

// ---------------------------------------------------------------------------
extern "C" void kernel_launch(void* const* d_in, const int* in_sizes, int n_in,
                              void* d_out, int out_size, void* d_ws, size_t ws_size,
                              hipStream_t stream) {
  (void)in_sizes; (void)n_in; (void)out_size; (void)ws_size;
  const float* x_r        = (const float*)d_in[0];
  const float* x_i        = (const float*)d_in[1];
  const float* w_qkv_r    = (const float*)d_in[2];
  const float* w_qkv_i    = (const float*)d_in[3];
  const float* b_qkv_r    = (const float*)d_in[4];
  const float* b_qkv_i    = (const float*)d_in[5];
  const float* w_out_r    = (const float*)d_in[6];
  const float* w_out_i    = (const float*)d_in[7];
  const float* b_out_r    = (const float*)d_in[8];
  const float* b_out_i    = (const float*)d_in[9];
  const float* w_layers_r = (const float*)d_in[10];
  const float* w_layers_i = (const float*)d_in[11];
  const float* b_layers_r = (const float*)d_in[12];
  const float* b_layers_i = (const float*)d_in[13];

  const size_t SZ_X    = (size_t)CM * CD;
  const size_t SZ_WQKV = (size_t)CNQKV * CD;
  const size_t SZ_WOUT = (size_t)CD * CHD;
  const size_t SZ_WLAY = (size_t)3 * CD * CD;
  const size_t SZ_HEAD = (size_t)CB * CH * CS * CD;
  const size_t SZ_SC   = (size_t)CB * CH * CS * CS;
  const size_t SZ_ATT  = (size_t)CM * CHD;
  const size_t SZ_LB   = (size_t)CM * CD;

  unsigned char* ws = (unsigned char*)d_ws;
  size_t off = 0;
  auto take = [&](size_t elems, size_t esz) {
    size_t o = off;
    off += ((elems * esz + 255) & ~(size_t)255);
    return o;
  };

  unsigned short* xbr = (unsigned short*)(ws + take(SZ_X, 2));
  unsigned short* xbi = (unsigned short*)(ws + take(SZ_X, 2));
  unsigned short* wqr = (unsigned short*)(ws + take(SZ_WQKV, 2));
  unsigned short* wqi = (unsigned short*)(ws + take(SZ_WQKV, 2));
  unsigned short* wor = (unsigned short*)(ws + take(SZ_WOUT, 2));
  unsigned short* woi = (unsigned short*)(ws + take(SZ_WOUT, 2));
  unsigned short* wlr = (unsigned short*)(ws + take(SZ_WLAY, 2));
  unsigned short* wli = (unsigned short*)(ws + take(SZ_WLAY, 2));
  unsigned short* qrb = (unsigned short*)(ws + take(SZ_HEAD, 2));
  unsigned short* qib = (unsigned short*)(ws + take(SZ_HEAD, 2));
  unsigned short* krb = (unsigned short*)(ws + take(SZ_HEAD, 2));
  unsigned short* kib = (unsigned short*)(ws + take(SZ_HEAD, 2));
  unsigned short* vtr = (unsigned short*)(ws + take(SZ_HEAD, 2));
  unsigned short* vti = (unsigned short*)(ws + take(SZ_HEAD, 2));
  float*          sc  = (float*)(ws + take(SZ_SC, 4));

  // Aliases: P (bf16, 64 MiB) over dead q/k buffers; attn + layer buffers over dead scores
  unsigned short* Pbf = qrb;
  unsigned char*  scb = (unsigned char*)sc;
  unsigned short* atr = (unsigned short*)scb;
  unsigned short* ati = (unsigned short*)(scb + SZ_ATT * 2);
  unsigned short* y0r = (unsigned short*)(scb + SZ_ATT * 4);
  unsigned short* y0i = y0r + SZ_LB;
  unsigned short* y1r = y0i + SZ_LB;
  unsigned short* y1i = y1r + SZ_LB;

  auto cast = [&](const float* src, unsigned short* dst, size_t n) {
    int blocks = (int)((n + 1023) / 1024);
    cast_bf16_kernel<<<dim3(blocks), 256, 0, stream>>>(src, dst, (int)n);
  };
  cast(x_r, xbr, SZ_X);        cast(x_i, xbi, SZ_X);
  cast(w_qkv_r, wqr, SZ_WQKV); cast(w_qkv_i, wqi, SZ_WQKV);
  cast(w_out_r, wor, SZ_WOUT); cast(w_out_i, woi, SZ_WOUT);
  cast(w_layers_r, wlr, SZ_WLAY); cast(w_layers_i, wli, SZ_WLAY);

  // 1) QKV: [4096,256] x [6144,256]^T complex, bias, scatter into q/k/v-transposed
  {
    EpiArgs ea{};
    ea.o0 = qrb; ea.o1 = qib; ea.o2 = krb; ea.o3 = kib; ea.o4 = vtr; ea.o5 = vti;
    ea.biasR = b_qkv_r; ea.biasI = b_qkv_i; ea.scale = 1.f;
    cgemm_wmma_kernel<0, 0><<<dim3(CNQKV / BN, CM / BM, 1), 256, 0, stream>>>(
        xbr, xbi, wqr, wqi, CD, CD, CD, 0, 0, ea);
  }
  // 2) scores: Re(q·conj(k)) / sqrt(D), batched over B*H
  {
    EpiArgs ea{};
    ea.o0 = sc; ea.scale = 1.f / 16.f;
    cgemm_wmma_kernel<1, 1><<<dim3(CS / BN, CS / BM, CB * CH), 256, 0, stream>>>(
        qrb, qib, krb, kib, CD, CD, CD, (long long)CS * CD, (long long)CS * CD, ea);
  }
  // 3) softmax rows -> P bf16
  softmax_kernel<<<dim3(CB * CH * CS), 256, 0, stream>>>(sc, Pbf);
  // 4) PV: P @ V (real x complex), batched; B operand = V^T (K-major contiguous)
  {
    EpiArgs ea{};
    ea.o0 = atr; ea.o1 = ati; ea.scale = 1.f;
    cgemm_wmma_kernel<2, 2><<<dim3(CD / BN, CS / BM, CB * CH), 256, 0, stream>>>(
        Pbf, nullptr, vtr, vti, CS, CS, CS, (long long)CS * CS, (long long)CD * CS, ea);
  }
  // 5) output projection: [4096,2048] x [256,2048]^T complex + bias
  {
    EpiArgs ea{};
    ea.o0 = y0r; ea.o1 = y0i; ea.biasR = b_out_r; ea.biasI = b_out_i; ea.scale = 1.f;
    cgemm_wmma_kernel<0, 3><<<dim3(CD / BN, CM / BM, 1), 256, 0, stream>>>(
        atr, ati, wor, woi, CHD, CHD, CHD, 0, 0, ea);
  }
  // 6) layer 0: CReLU(y0 @ W0^T + b0) -> y1
  {
    EpiArgs ea{};
    ea.o0 = y1r; ea.o1 = y1i; ea.biasR = b_layers_r; ea.biasI = b_layers_i; ea.scale = 1.f;
    cgemm_wmma_kernel<0, 4><<<dim3(CD / BN, CM / BM, 1), 256, 0, stream>>>(
        y0r, y0i, wlr, wli, CD, CD, CD, 0, 0, ea);
  }
  // 7) layer 1 -> y0
  {
    EpiArgs ea{};
    ea.o0 = y0r; ea.o1 = y0i; ea.biasR = b_layers_r + CD; ea.biasI = b_layers_i + CD; ea.scale = 1.f;
    cgemm_wmma_kernel<0, 4><<<dim3(CD / BN, CM / BM, 1), 256, 0, stream>>>(
        y1r, y1i, wlr + (size_t)CD * CD, wli + (size_t)CD * CD, CD, CD, CD, 0, 0, ea);
  }
  // 8) layer 2 -> d_out f32 [B,S,D,2]
  {
    EpiArgs ea{};
    ea.o0 = d_out; ea.biasR = b_layers_r + 2 * CD; ea.biasI = b_layers_i + 2 * CD; ea.scale = 1.f;
    cgemm_wmma_kernel<0, 5><<<dim3(CD / BN, CM / BM, 1), 256, 0, stream>>>(
        y0r, y0i, wlr + (size_t)2 * CD * CD, wli + (size_t)2 * CD * CD, CD, CD, CD, 0, 0, ea);
  }
}
